// LiZAttention_80307298501090
// MI455X (gfx1250) — matrix operand order repaired and checked by reading, not compiled, not run on hardware
//
#include <hip/hip_runtime.h>
#include <hip/hip_bf16.h>

#define NTOK   2048
#define HIDDIM 2048
#define NHEADS 16
#define NKVH   4
#define HDIM   128
#define CHUNK_ 64
#define NCHUNK 32

typedef __attribute__((ext_vector_type(16))) __bf16 v16bf;
typedef __attribute__((ext_vector_type(8)))  float  v8f;
typedef int v4i_vs __attribute__((vector_size(16)));   // matches builtin's pointee type

union FragU { v16bf v; uint4 q[2]; };
union Pack8 { uint4 q; __bf16 b[8]; };

// ---- CDNA5 async global->LDS copy (ASYNCcnt), with safe fallback ----
#if defined(__has_builtin)
#if __has_builtin(__builtin_amdgcn_global_load_async_to_lds_b128) && \
    __has_builtin(__builtin_amdgcn_s_wait_asynccnt)
#define HAVE_ASYNC_LDS 1
#endif
#endif

__device__ __forceinline__ void async_cp16(void* lds_dst, const void* gsrc) {
#ifdef HAVE_ASYNC_LDS
  __builtin_amdgcn_global_load_async_to_lds_b128(
      (__attribute__((address_space(1))) v4i_vs*)(gsrc),
      (__attribute__((address_space(3))) v4i_vs*)(lds_dst), 0, 0);
#else
  *(uint4*)lds_dst = *(const uint4*)gsrc;
#endif
}
__device__ __forceinline__ void async_join() {
#ifdef HAVE_ASYNC_LDS
  __builtin_amdgcn_s_wait_asynccnt(0);
#endif
}

__device__ __forceinline__ v8f wmma_bf16(v16bf a, v16bf b, v8f c) {
  return __builtin_amdgcn_wmma_f32_16x16x32_bf16(false, a, false, b, (short)0, c, false, false);
}

// A fragment: rows m0..m0+15 of a row-major [*, ld] bf16 matrix, K window kk..kk+31.
// ISA layout: lane->M (lane&15), halves hold K {0..7,16..23} / {8..15,24..31}.
__device__ __forceinline__ v16bf fragA(const __bf16* base, int m0, int ld, int kk) {
  int lane = threadIdx.x & 31;
  const __bf16* p = base + (size_t)(m0 + (lane & 15)) * ld + kk + ((lane >> 4) << 3);
  FragU f;
  f.q[0] = *(const uint4*)(p);
  f.q[1] = *(const uint4*)(p + 16);
  return f.v;
}

// B fragment: source stored K-major [k][n] (n contiguous), cols n0..n0+15, K kk..kk+31.
// ISA layout: lane->N (lane&15), halves hold K {0..7,16..23} / {8..15,24..31}.
__device__ __forceinline__ v16bf fragB(const __bf16* base, int n0, int ld, int kk) {
  int lane = threadIdx.x & 31;
  const __bf16* p = base + (size_t)(kk + ((lane >> 4) << 3)) * ld + n0 + (lane & 15);
  v16bf v;
#pragma unroll
  for (int s = 0; s < 8; ++s) { v[s] = p[(size_t)s * ld]; v[s + 8] = p[(size_t)(s + 16) * ld]; }
  return v;
}

// ---------------------------------------------------------------- casts
__global__ void cast_f32_bf16(const float* __restrict__ src, __bf16* __restrict__ dst, int n4) {
  int i = blockIdx.x * blockDim.x + threadIdx.x;
  if (i >= n4) return;
  float4 x = ((const float4*)src)[i];
  __bf16* o = dst + (size_t)i * 4;
  o[0] = (__bf16)x.x; o[1] = (__bf16)x.y; o[2] = (__bf16)x.z; o[3] = (__bf16)x.w;
}

// ---------------------------------------------------------------- GEMM: C[M,Nf] = A[M,K] * W[Nf,K]^T
// Vectorized staging: one b128 load per thread (K-contiguous in W), 8 ds_store_b16 scatter.
__device__ __forceinline__ void stage_wt(__bf16* Wt, const __bf16* __restrict__ W,
                                         int n0, int K, int kk) {
  int t = threadIdx.x;
  int n = t >> 2, kq = (t & 3) * 8;                  // 64 rows x 4 b128-chunks
  Pack8 u;
  u.q = *(const uint4*)(W + (size_t)(n0 + n) * K + kk + kq);
#pragma unroll
  for (int j = 0; j < 8; ++j) Wt[(kq + j) * 64 + n] = u.b[j];
}

__global__ void __launch_bounds__(256)
gemm_bf16_tn(const __bf16* __restrict__ A, const __bf16* __restrict__ W,
             float* __restrict__ C, int M, int Nf, int K) {
  __shared__ __bf16 Wt[2][32 * 64];                  // double-buffered W^T tile, [k][n]
  int tid = threadIdx.x, wave = tid >> 5, lane = tid & 31;
  int n0 = blockIdx.x * 64;
  int m0 = blockIdx.y * 256 + wave * 32;
  v8f acc[2][4];
  for (int i = 0; i < 2; ++i) for (int j = 0; j < 4; ++j) acc[i][j] = (v8f)0.0f;

  stage_wt(Wt[0], W, n0, K, 0);
  int buf = 0;
  for (int kk = 0; kk < K; kk += 32) {
    __syncthreads();                                 // publish Wt[buf]
    if (kk + 32 < K) {
      stage_wt(Wt[buf ^ 1], W, n0, K, kk + 32);      // overlap next tile with WMMAs
      __builtin_prefetch(A + (size_t)m0 * K + kk + 32, 0, 1);
    }
    v16bf a0 = fragA(A, m0,      K, kk);
    v16bf a1 = fragA(A, m0 + 16, K, kk);
#pragma unroll
    for (int j = 0; j < 4; ++j) {
      v16bf b = fragB(Wt[buf], j * 16, 64, 0);
      acc[0][j] = wmma_bf16(a0, b, acc[0][j]);
      acc[1][j] = wmma_bf16(a1, b, acc[1][j]);
    }
    buf ^= 1;
  }
  int half = lane >> 4, col = lane & 15;
  for (int i = 0; i < 2; ++i)
    for (int j = 0; j < 4; ++j)
#pragma unroll
      for (int r = 0; r < 8; ++r)
        C[(size_t)(m0 + i * 16 + r + 8 * half) * Nf + n0 + j * 16 + col] = acc[i][j][r];
}

// ---------------------------------------------------------------- q feature map (softmax over HD)
__global__ void qfeat_kernel(const float* __restrict__ q,
                             __bf16* __restrict__ ql, __bf16* __restrict__ qr) {
  int row  = blockIdx.x * 8 + (threadIdx.x >> 5);    // h*NTOK + m
  int lane = threadIdx.x & 31;
  int h = row >> 11, m = row & 2047;
  const float* src = q + (size_t)m * (NHEADS * HDIM) + h * HDIM + lane * 4;
  float4 x = *(const float4*)src;
  float xa[4] = {x.x, x.y, x.z, x.w};
  float mx = fmaxf(fmaxf(xa[0], xa[1]), fmaxf(xa[2], xa[3]));
  for (int off = 16; off; off >>= 1) mx = fmaxf(mx, __shfl_xor(mx, off));
  float e[4], s = 0.f;
#pragma unroll
  for (int i = 0; i < 4; ++i) { e[i] = expf(xa[i] - mx); s += e[i]; }
  for (int off = 16; off; off >>= 1) s += __shfl_xor(s, off);
  float inv = 1.f / s;
  size_t base = ((size_t)h * NTOK + m) * HDIM + lane * 4;
#pragma unroll
  for (int i = 0; i < 4; ++i) { ql[base + i] = (__bf16)(e[i] * inv); qr[base + i] = (__bf16)xa[i]; }
}

// ---------------------------------------------------------------- k/v feature maps + gate, GQA expand
__global__ void kvfeat_kernel(const float* __restrict__ k, const float* __restrict__ v,
                              __bf16* __restrict__ kl, __bf16* __restrict__ kr,
                              __bf16* __restrict__ kb, __bf16* __restrict__ vb,
                              __bf16* __restrict__ vr) {
  int row  = blockIdx.x * 8 + (threadIdx.x >> 5);    // m*NKVH + hk
  int lane = threadIdx.x & 31;
  int hk = row & 3, m = row >> 2;
  const float* ks = k + (size_t)m * (NKVH * HDIM) + hk * HDIM + lane * 4;
  const float* vs = v + (size_t)m * (NKVH * HDIM) + hk * HDIM + lane * 4;
  float4 x = *(const float4*)ks;
  float4 y = *(const float4*)vs;
  float xa[4] = {x.x, x.y, x.z, x.w}, ya[4] = {y.x, y.y, y.z, y.w};
  float mx = fmaxf(fmaxf(xa[0], xa[1]), fmaxf(xa[2], xa[3]));
  for (int off = 16; off; off >>= 1) mx = fmaxf(mx, __shfl_xor(mx, off));
  float e[4], s = 0.f;
#pragma unroll
  for (int i = 0; i < 4; ++i) { e[i] = expf(xa[i] - mx); s += e[i]; }
  for (int off = 16; off; off >>= 1) s += __shfl_xor(s, off);
  float inv = 1.f / s;
  float klv[4], kbv[4], vbv[4];
#pragma unroll
  for (int i = 0; i < 4; ++i) {
    klv[i] = e[i] * inv;
    float g = (fminf(xa[i], 0.f) - log1pf(expf(-fabsf(xa[i])))) * 0.0625f;  // logsigmoid/16
    kbv[i] = klv[i] * g;
    vbv[i] = ya[i] * g;
  }
  for (int hg = 0; hg < 4; ++hg) {                   // repeat_kv: h = hk*GROUPS + hg
    int h = hk * 4 + hg;
    size_t base = ((size_t)h * NTOK + m) * HDIM + lane * 4;
#pragma unroll
    for (int i = 0; i < 4; ++i) {
      kl[base + i] = (__bf16)klv[i]; kr[base + i] = (__bf16)xa[i];
      kb[base + i] = (__bf16)kbv[i]; vb[base + i] = (__bf16)vbv[i];
      vr[base + i] = (__bf16)ya[i];
    }
  }
}

// vectorized transpose stage: 64xHDIM row-major global -> [d][t] LDS (b128 load + 8 b16 scatter)
__device__ __forceinline__ void stage_transpose64(__bf16* dstT, const __bf16* __restrict__ src) {
  int tid = threadIdx.x;
#pragma unroll
  for (int it = 0; it < 4; ++it) {
    int idx = tid + it * 256;                        // 0..1023 b128-chunks
    int t = idx >> 4, dq = (idx & 15) * 8;
    Pack8 u;
    u.q = *(const uint4*)(src + t * HDIM + dq);
#pragma unroll
    for (int j = 0; j < 8; ++j) dstT[(dq + j) * 64 + t] = u.b[j];
  }
}

// ---------------------------------------------------------------- chunked delta-rule scan (1 head / block)
__global__ void __launch_bounds__(256)
delta_kernel(const __bf16* __restrict__ qlg, const __bf16* __restrict__ klg,
             const __bf16* __restrict__ kbg, const __bf16* __restrict__ vbg,
             float* __restrict__ olin) {
  __shared__ float  Sf[128 * 128];     // f32 recurrent state (64KB)
  __shared__ __bf16 Sb[128 * 128];     // bf16 shadow, row-major [k][n]
  __shared__ __bf16 qt [64 * 128];
  __shared__ __bf16 kT [128 * 64];     // kl^T  [d][t]
  __shared__ __bf16 kbt[64 * 128];
  __shared__ __bf16 vbt[64 * 128];
  __shared__ float  Af [64 * 64];      // unit-lower A for triangular solve
  __shared__ __bf16 wt [64 * 128];     // holds -w (u_i = u + (-w)@S fits WMMA D=AB+C)
  __shared__ __bf16 ut [64 * 128];     // u, then u_i
  __shared__ __bf16 Pt [64 * 64];      // causal attn (bf16)
  int h = blockIdx.x, tid = threadIdx.x, wave = tid >> 5, lane = tid & 31;
  int half = lane >> 4, col = lane & 15;
  const __bf16* qlh = qlg + (size_t)h * NTOK * HDIM;
  const __bf16* klh = klg + (size_t)h * NTOK * HDIM;
  const __bf16* kbh = kbg + (size_t)h * NTOK * HDIM;
  const __bf16* vbh = vbg + (size_t)h * NTOK * HDIM;
  for (int i = tid; i < 128 * 128; i += 256) { Sf[i] = 0.f; Sb[i] = (__bf16)0.f; }
  __syncthreads();

  for (int c = 0; c < NCHUNK; ++c) {
    size_t t0 = (size_t)c * CHUNK_ * HDIM;
    for (int i = tid; i < CHUNK_ * HDIM / 8; i += 256) {   // async row-major tiles
      async_cp16(((uint4*)qt ) + i, ((const uint4*)(qlh + t0)) + i);
      async_cp16(((uint4*)kbt) + i, ((const uint4*)(kbh + t0)) + i);
      async_cp16(((uint4*)vbt) + i, ((const uint4*)(vbh + t0)) + i);
    }
    stage_transpose64(kT, klh + t0);                       // kl^T for B/A operands
    async_join();
    __syncthreads();
    // (1) A = I + strict_tril(kb @ kl^T)
    for (int rep = 0; rep < 2; ++rep) {
      int tile = wave * 2 + rep, tm = (tile >> 2) * 16, tn = (tile & 3) * 16;
      v8f a = (v8f)0.0f;
      for (int kk = 0; kk < 128; kk += 32)
        a = wmma_bf16(fragA(kbt, tm, 128, kk), fragB(kT, tn, 64, kk), a);
#pragma unroll
      for (int r = 0; r < 8; ++r) {
        int rr = tm + r + 8 * half, cc = tn + col;
        Af[rr * 64 + cc] = (cc < rr) ? a[r] : (cc == rr ? 1.f : 0.f);
      }
    }
    __syncthreads();
    // (2) forward substitution: wt = -(A^-1 kb), ut = A^-1 vb  (columns thread-private)
    {
      int cc = tid & 127;
      const __bf16* src = (tid < 128) ? kbt : vbt;
      __bf16* dst = (tid < 128) ? wt : ut;
      float sgn = (tid < 128) ? -1.f : 1.f;
      for (int i = 0; i < 64; ++i) {
        float acc = sgn * (float)src[i * 128 + cc];
        for (int j = 0; j < i; ++j)
          acc -= Af[i * 64 + j] * (float)dst[j * 128 + cc];
        dst[i * 128 + cc] = (__bf16)acc;
      }
    }
    __syncthreads();
    // (3) u_i = u + (-w) @ S   (C loaded from ut, D overwrites same tile)
    for (int rep = 0; rep < 4; ++rep) {
      int tile = wave * 4 + rep, tm = (tile >> 3) * 16, tn = (tile & 7) * 16;
      v8f a;
#pragma unroll
      for (int r = 0; r < 8; ++r) a[r] = (float)ut[(tm + r + 8 * half) * 128 + tn + col];
      for (int kk = 0; kk < 128; kk += 32)
        a = wmma_bf16(fragA(wt, tm, 128, kk), fragB(Sb, tn, 128, kk), a);
#pragma unroll
      for (int r = 0; r < 8; ++r) ut[(tm + r + 8 * half) * 128 + tn + col] = (__bf16)a[r];
    }
    // (4) P = tril(ql @ kl^T)
    for (int rep = 0; rep < 2; ++rep) {
      int tile = wave * 2 + rep, tm = (tile >> 2) * 16, tn = (tile & 3) * 16;
      v8f a = (v8f)0.0f;
      for (int kk = 0; kk < 128; kk += 32)
        a = wmma_bf16(fragA(qt, tm, 128, kk), fragB(kT, tn, 64, kk), a);
#pragma unroll
      for (int r = 0; r < 8; ++r) {
        int rr = tm + r + 8 * half, cc = tn + col;
        Pt[rr * 64 + cc] = (cc <= rr) ? (__bf16)a[r] : (__bf16)0.f;
      }
    }
    __syncthreads();
    // (5) o = ql @ S + P @ u_i  -> global o_lin (head-major)
    for (int rep = 0; rep < 4; ++rep) {
      int tile = wave * 4 + rep, tm = (tile >> 3) * 16, tn = (tile & 7) * 16;
      v8f a = (v8f)0.0f;
      for (int kk = 0; kk < 128; kk += 32)
        a = wmma_bf16(fragA(qt, tm, 128, kk), fragB(Sb, tn, 128, kk), a);
      for (int kk = 0; kk < 64; kk += 32)
        a = wmma_bf16(fragA(Pt, tm, 64, kk), fragB(ut, tn, 128, kk), a);
#pragma unroll
      for (int r = 0; r < 8; ++r)
        olin[(size_t)h * NTOK * HDIM + (size_t)(c * CHUNK_ + tm + r + 8 * half) * HDIM + tn + col] = a[r];
    }
    // (6) S += kl^T @ u_i  (f32 accumulate in LDS)
    for (int rep = 0; rep < 8; ++rep) {
      int tile = wave * 8 + rep, tm = (tile >> 3) * 16, tn = (tile & 7) * 16;
      v8f a;
#pragma unroll
      for (int r = 0; r < 8; ++r) a[r] = Sf[(tm + r + 8 * half) * 128 + tn + col];
      for (int kk = 0; kk < 64; kk += 32)
        a = wmma_bf16(fragA(kT, tm, 64, kk), fragB(ut, tn, 128, kk), a);
#pragma unroll
      for (int r = 0; r < 8; ++r) Sf[(tm + r + 8 * half) * 128 + tn + col] = a[r];
    }
    __syncthreads();
    for (int i = tid; i < 128 * 128; i += 256) Sb[i] = (__bf16)Sf[i];
    __syncthreads();
  }
}

// ---------------------------------------------------------------- flash causal softmax attention
__global__ void __launch_bounds__(256)
flash_kernel(const __bf16* __restrict__ qrg, const __bf16* __restrict__ krg,
             const __bf16* __restrict__ vrg, float* __restrict__ obase) {
  __shared__ __bf16 qtile[64 * 128];
  __shared__ __bf16 kTt  [128 * 64];   // k^T tile [d][t]
  __shared__ __bf16 vtile[64 * 128];
  __shared__ float  Stile[64 * 64];
  __shared__ __bf16 Ptile[64 * 64];
  __shared__ float  Ot   [64 * 128];
  __shared__ float  mrow[64], lrow[64], arow[64];
  int qb = blockIdx.x, h = blockIdx.y;
  int tid = threadIdx.x, wave = tid >> 5, lane = tid & 31, half = lane >> 4, col = lane & 15;
  const __bf16* qh = qrg + (size_t)h * NTOK * HDIM;
  const __bf16* kh = krg + (size_t)h * NTOK * HDIM;
  const __bf16* vh = vrg + (size_t)h * NTOK * HDIM;
  for (int i = tid; i < 64 * 16; i += 256)
    async_cp16(((uint4*)qtile) + i, ((const uint4*)(qh + (size_t)qb * 8192)) + i);
  for (int i = tid; i < 8192; i += 256) Ot[i] = 0.f;
  if (tid < 64) { mrow[tid] = -3.0e38f; lrow[tid] = 0.f; }
  async_join();
  __syncthreads();
  const float scale = 0.08838834764831845f;          // 128^-0.5

  for (int kb = 0; kb <= qb; ++kb) {
    for (int i = tid; i < 64 * 16; i += 256)
      async_cp16(((uint4*)vtile) + i, ((const uint4*)(vh + (size_t)kb * 8192)) + i);
    stage_transpose64(kTt, kh + (size_t)kb * 8192);
    async_join();
    __syncthreads();
    // S = scale * q @ k^T (causal mask on diagonal block)
    for (int rep = 0; rep < 2; ++rep) {
      int tile = wave * 2 + rep, tm = (tile >> 2) * 16, tn = (tile & 3) * 16;
      v8f a = (v8f)0.0f;
      for (int kk = 0; kk < 128; kk += 32)
        a = wmma_bf16(fragA(qtile, tm, 128, kk), fragB(kTt, tn, 64, kk), a);
#pragma unroll
      for (int r = 0; r < 8; ++r) {
        int rr = tm + r + 8 * half, cc = tn + col;
        bool keep = (kb < qb) || (cc <= rr);
        Stile[rr * 64 + cc] = keep ? a[r] * scale : -3.0e38f;
      }
    }
    __syncthreads();
    if (tid < 64) {                                   // online softmax, one row per thread
      float mold = mrow[tid], mx = mold;
      for (int cc = 0; cc < 64; ++cc) mx = fmaxf(mx, Stile[tid * 64 + cc]);
      float alpha = expf(mold - mx), s = 0.f;
      for (int cc = 0; cc < 64; ++cc) {
        float p = expf(Stile[tid * 64 + cc] - mx);
        Ptile[tid * 64 + cc] = (__bf16)p; s += p;
      }
      lrow[tid] = lrow[tid] * alpha + s; mrow[tid] = mx; arow[tid] = alpha;
    }
    __syncthreads();
    for (int i = tid; i < 8192; i += 256) Ot[i] *= arow[i >> 7];
    __syncthreads();
    // O += P @ V
    for (int rep = 0; rep < 4; ++rep) {
      int tile = wave * 4 + rep, tm = (tile >> 3) * 16, tn = (tile & 7) * 16;
      v8f a;
#pragma unroll
      for (int r = 0; r < 8; ++r) a[r] = Ot[(tm + r + 8 * half) * 128 + tn + col];
      for (int kk = 0; kk < 64; kk += 32)
        a = wmma_bf16(fragA(Ptile, tm, 64, kk), fragB(vtile, tn, 128, kk), a);
#pragma unroll
      for (int r = 0; r < 8; ++r) Ot[(tm + r + 8 * half) * 128 + tn + col] = a[r];
    }
    __syncthreads();
  }
  for (int i = tid; i < 8192; i += 256)
    obase[(size_t)h * NTOK * HDIM + (size_t)qb * 8192 + i] = Ot[i] / lrow[i >> 7];
}

// ---------------------------------------------------------------- blend (head-major -> token-major bf16)
__global__ void combine_kernel(const float* __restrict__ ol, const float* __restrict__ ob,
                               __bf16* __restrict__ oc) {
  size_t i = (size_t)blockIdx.x * 256 + threadIdx.x;  // over NHEADS*NTOK*HDIM
  int h = (int)(i >> 18);
  int rem = (int)(i & 262143), m = rem >> 7, d = rem & 127;
  oc[(size_t)m * 2048 + h * 128 + d] = (__bf16)(0.5f * (ol[i] + ob[i]));
}

// ---------------------------------------------------------------- launch
extern "C" void kernel_launch(void* const* d_in, const int* in_sizes, int n_in,
                              void* d_out, int out_size, void* d_ws, size_t ws_size,
                              hipStream_t stream) {
  const float* hidden = (const float*)d_in[0];
  const float* Wq = (const float*)d_in[1];
  const float* Wk = (const float*)d_in[2];
  const float* Wv = (const float*)d_in[3];
  const float* Wo = (const float*)d_in[4];
  float* out = (float*)d_out;

  char* ws = (char*)d_ws;
  size_t off = 0;
  auto take = [&](size_t bytes) -> char* {
    char* p = ws + off;
    off += (bytes + 255) & ~(size_t)255;
    return p;
  };
  __bf16* hB   = (__bf16*)take(4194304ull * 2);
  __bf16* wqB  = (__bf16*)take(4194304ull * 2);
  __bf16* wkB  = (__bf16*)take(1048576ull * 2);
  __bf16* wvB  = (__bf16*)take(1048576ull * 2);
  __bf16* woB  = (__bf16*)take(4194304ull * 2);
  float*  qF   = (float*) take(4194304ull * 4);
  float*  kF   = (float*) take(1048576ull * 4);
  float*  vF   = (float*) take(1048576ull * 4);
  __bf16* qlB  = (__bf16*)take(4194304ull * 2);
  __bf16* qrB  = (__bf16*)take(4194304ull * 2);
  __bf16* klB  = (__bf16*)take(4194304ull * 2);
  __bf16* krB  = (__bf16*)take(4194304ull * 2);
  __bf16* kbB  = (__bf16*)take(4194304ull * 2);
  __bf16* vbB  = (__bf16*)take(4194304ull * 2);
  __bf16* vrB  = (__bf16*)take(4194304ull * 2);
  float*  olin = (float*) take(4194304ull * 4);
  float*  obas = (float*) take(4194304ull * 4);
  __bf16* ocB  = (__bf16*)take(4194304ull * 2);

  cast_f32_bf16<<<4096, 256, 0, stream>>>(hidden, hB, 1048576);
  cast_f32_bf16<<<4096, 256, 0, stream>>>(Wq, wqB, 1048576);
  cast_f32_bf16<<<1024, 256, 0, stream>>>(Wk, wkB, 262144);
  cast_f32_bf16<<<1024, 256, 0, stream>>>(Wv, wvB, 262144);
  cast_f32_bf16<<<4096, 256, 0, stream>>>(Wo, woB, 1048576);

  gemm_bf16_tn<<<dim3(32, 8), 256, 0, stream>>>(hB, wqB, qF, 2048, 2048, 2048);
  gemm_bf16_tn<<<dim3(8, 8),  256, 0, stream>>>(hB, wkB, kF, 2048, 512, 2048);
  gemm_bf16_tn<<<dim3(8, 8),  256, 0, stream>>>(hB, wvB, vF, 2048, 512, 2048);

  qfeat_kernel<<<4096, 256, 0, stream>>>(qF, qlB, qrB);
  kvfeat_kernel<<<1024, 256, 0, stream>>>(kF, vF, klB, krB, kbB, vbB, vrB);

  delta_kernel<<<16, 256, 0, stream>>>(qlB, klB, kbB, vbB, olin);
  flash_kernel<<<dim3(32, 16), 256, 0, stream>>>(qrB, krB, vrB, obas);

  combine_kernel<<<16384, 256, 0, stream>>>(olin, obas, ocB);
  gemm_bf16_tn<<<dim3(32, 8), 256, 0, stream>>>(ocB, woB, out, 2048, 2048, 2048);
}
